// MultiHeadAttention_16475494548106
// MI455X (gfx1250) — matrix operand order repaired and checked by reading, not compiled, not run on hardware
//
#include <hip/hip_runtime.h>

#define DIM  1024
#define SEQ  2048
#define NH   16
#define HD   64
#define NTOK 4096   // B*S

// LDS staging geometry for attention (per 32-key block)
#define K_ROWB   144                 // 64 bf16 = 128B + 16B pad
#define V_ROWB   80                  // 32 bf16 = 64B + 16B pad
#define K_TILEB  (32 * K_ROWB)       // 4608
#define V_TILEB  (64 * V_ROWB)       // 5120
#define BUF_B    (K_TILEB + V_TILEB) // 9728
#define XPOSE_B  (8 * 16 * HD * 2)   // 16384, o^T transpose scratch

typedef __bf16 bf16;
typedef __attribute__((ext_vector_type(16))) __bf16 v16bf;
typedef __attribute__((ext_vector_type(8)))  __bf16 v8bf;
typedef __attribute__((ext_vector_type(4)))  __bf16 v4bf;
typedef __attribute__((ext_vector_type(8)))  float  v8f;

static __device__ __forceinline__ v8f wmma_bf(v16bf a, v16bf b, v8f c) {
  return __builtin_amdgcn_wmma_f32_16x16x32_bf16(false, a, false, b, (short)0, c,
                                                 false, false);
}

// A-fragment (16x32, M rows): lane L holds row M=L%16, K=(L>>4)*16+{0..15}.
static __device__ __forceinline__ v16bf afrag(const bf16* p) {
  v8bf lo = *(const v8bf*)p;
  v8bf hi = *(const v8bf*)(p + 8);
  v16bf r;
#pragma unroll
  for (int i = 0; i < 8; ++i) { r[i] = lo[i]; r[8 + i] = hi[i]; }
  return r;
}

// B-fragment (32x16, N cols): lane L holds col N=L%16, K=(L>>4)*16+{0..15}.
static __device__ __forceinline__ v16bf bfrag(const bf16* p) {
  v8bf lo = *(const v8bf*)p;
  v8bf hi = *(const v8bf*)(p + 8);
  v16bf r;
#pragma unroll
  for (int i = 0; i < 8; ++i) { r[i] = lo[i]; r[8 + i] = hi[i]; }
  return r;
}

// Low 32 bits of a generic pointer into the LDS aperture == LDS byte offset
// (ISA 10.2).
static __device__ __forceinline__ unsigned lds_off(const void* p) {
  return (unsigned)(unsigned long long)p;
}

// One 16B async global->LDS copy per lane (GVS mode), tracked by ASYNCcnt.
static __device__ __forceinline__ void async_cp16(unsigned lds_byte,
                                                  const void* saddr,
                                                  unsigned voff) {
  asm volatile("global_load_async_to_lds_b128 %0, %1, %2"
               :: "v"(lds_byte), "v"(voff), "s"(saddr) : "memory");
}

static __device__ __forceinline__ void wait_async0() {
  asm volatile("s_wait_asynccnt 0x0" ::: "memory");
}
static __device__ __forceinline__ void wait_ds0() {
  asm volatile("s_wait_dscnt 0x0" ::: "memory");
}

// ---------------------------------------------------------------- fp32 -> bf16
__global__ __launch_bounds__(256) void cvt_bf16(const float* __restrict__ in,
                                                bf16* __restrict__ out, int n) {
  const int i = (blockIdx.x * 256 + threadIdx.x) * 4;
  if (i < n) {
    const float4 v = *(const float4*)(in + i);
    v4bf r = {(bf16)v.x, (bf16)v.y, (bf16)v.z, (bf16)v.w};
    *(v4bf*)(out + i) = r;
  }
}

// ------------------------------------------------- projection GEMM (bf16 out)
// y = X @ W^T + b.  Wave tile 32x64 (2 M-tiles reuse each B-frag), 8 waves ->
// WG tile 64x256.  vmode==0: out[b,h,s,d].  vmode==1: out[b,h,d,s] (V^T).
__global__ __launch_bounds__(256) void gemm_proj(
    const bf16* __restrict__ X, const bf16* __restrict__ W,
    const float* __restrict__ bias, bf16* __restrict__ out, int vmode) {
  const int wid = threadIdx.x >> 5, lane = threadIdx.x & 31;
  const int lr = lane & 15, lh = lane >> 4;
  const int r0 = blockIdx.x * 64 + (wid >> 2) * 32;
  const int c0 = blockIdx.y * 256 + (wid & 3) * 64;
  v8f acc[2][4] = {};
  const bf16* xp = X + (r0 + lr) * DIM + lh * 16;
  const bf16* wp = W + (c0 + lr) * DIM + lh * 16;
#pragma unroll 2
  for (int k0 = 0; k0 < DIM; k0 += 32) {
    const v16bf a0 = afrag(xp + k0);
    const v16bf a1 = afrag(xp + 16 * DIM + k0);
#pragma unroll
    for (int t = 0; t < 4; ++t) {
      const v16bf b = bfrag(wp + t * (16 * DIM) + k0);
      acc[0][t] = wmma_bf(a0, b, acc[0][t]);
      acc[1][t] = wmma_bf(a1, b, acc[1][t]);
    }
  }
#pragma unroll
  for (int t = 0; t < 4; ++t) {
    const int col = c0 + t * 16 + lr;
    const float bv = bias[col];
    const int h = col >> 6, dd = col & (HD - 1);
#pragma unroll
    for (int r = 0; r < 2; ++r)
#pragma unroll
      for (int i = 0; i < 8; ++i) {
        const int row = r0 + r * 16 + i + lh * 8;
        const int bb = row >> 11, s = row & (SEQ - 1);
        const bf16 v = (bf16)(acc[r][t][i] + bv);
        if (vmode == 0)
          out[((bb * NH + h) * SEQ + s) * HD + dd] = v;
        else
          out[((bb * NH + h) * HD + dd) * SEQ + s] = v;
      }
  }
}

// ------------------------------------------------- output GEMM (fp32 out)
__global__ __launch_bounds__(256) void gemm_out(
    const bf16* __restrict__ X, const bf16* __restrict__ W,
    const float* __restrict__ bias, float* __restrict__ out) {
  const int wid = threadIdx.x >> 5, lane = threadIdx.x & 31;
  const int lr = lane & 15, lh = lane >> 4;
  const int r0 = blockIdx.x * 64 + (wid >> 2) * 32;
  const int c0 = blockIdx.y * 256 + (wid & 3) * 64;
  v8f acc[2][4] = {};
  const bf16* xp = X + (r0 + lr) * DIM + lh * 16;
  const bf16* wp = W + (c0 + lr) * DIM + lh * 16;
#pragma unroll 2
  for (int k0 = 0; k0 < DIM; k0 += 32) {
    const v16bf a0 = afrag(xp + k0);
    const v16bf a1 = afrag(xp + 16 * DIM + k0);
#pragma unroll
    for (int t = 0; t < 4; ++t) {
      const v16bf b = bfrag(wp + t * (16 * DIM) + k0);
      acc[0][t] = wmma_bf(a0, b, acc[0][t]);
      acc[1][t] = wmma_bf(a1, b, acc[1][t]);
    }
  }
#pragma unroll
  for (int t = 0; t < 4; ++t) {
    const int col = c0 + t * 16 + lr;
    const float bv = bias[col];
#pragma unroll
    for (int r = 0; r < 2; ++r)
#pragma unroll
      for (int i = 0; i < 8; ++i)
        out[(r0 + r * 16 + i + lh * 8) * DIM + col] = acc[r][t][i] + bv;
  }
}

// Stage one 32-key block (K: 32x64 bf16, V^T: 64x32 bf16) into LDS.
// 8 waves x 32 lanes x 16B = 4KB per tile; each wave copies disjoint slices.
static __device__ __forceinline__ void stage_block(const bf16* kb, const bf16* vb,
                                                   char* sm, int j0,
                                                   int wid, int lane) {
  const int flat = wid * 512 + lane * 16;
  {  // K tile: row stride 128B in global, padded to 144B in LDS
    const int row = flat >> 7, colb = flat & 127;
    async_cp16(lds_off(sm + row * K_ROWB + colb), kb,
               (unsigned)((j0 + row) * 128 + colb));
  }
  {  // V^T tile: row stride SEQ*2 B in global, 64B slice -> 80B rows in LDS
    const int row = flat >> 6, colb = flat & 63;
    async_cp16(lds_off(sm + K_TILEB + row * V_ROWB + colb), vb,
               (unsigned)(row * (SEQ * 2) + j0 * 2 + colb));
  }
}

// ------------------------------------------------- flash attention
// One wave per 16 queries (8 waves = 128 per WG).  K/V^T blocks staged into
// double-buffered LDS with async copies overlapping WMMA+softmax.
__global__ __launch_bounds__(256) void attn_fwd(
    const bf16* __restrict__ Q, const bf16* __restrict__ K,
    const bf16* __restrict__ VT, bf16* __restrict__ O) {
  __shared__ __align__(16) char smem[2 * BUF_B + XPOSE_B];
  const int wid = threadIdx.x >> 5, lane = threadIdx.x & 31;
  const int lr = lane & 15, lh = lane >> 4;
  const int b = blockIdx.z, h = blockIdx.y;
  const int m0 = blockIdx.x * 128 + wid * 16;
  const bf16* qb = Q + (b * NH + h) * SEQ * HD;
  const bf16* kb = K + (b * NH + h) * SEQ * HD;
  const bf16* vb = VT + (b * NH + h) * HD * SEQ;

  // q^T B-fragments (col m = q row), live across the whole loop
  const v16bf qf0 = bfrag(qb + (m0 + lr) * HD + lh * 16);
  const v16bf qf1 = bfrag(qb + (m0 + lr) * HD + 32 + lh * 16);

  v8f o0 = {}, o1 = {}, o2 = {}, o3 = {};
  float mrun = -1e30f, lrun = 0.f;
  const float sc = 0.18033688f;  // log2(e)/sqrt(64)

  stage_block(kb, vb, smem, 0, wid, lane);
  wait_async0();
  __syncthreads();
  int cur = 0;

  for (int j0 = 0; j0 < SEQ; j0 += 32) {
    if (j0 + 32 < SEQ)                     // prefetch next block
      stage_block(kb, vb, smem + (cur ^ 1) * BUF_B, j0 + 32, wid, lane);

    const bf16* kt = (const bf16*)(smem + cur * BUF_B);
    const bf16* vt = (const bf16*)(smem + cur * BUF_B + K_TILEB);

    // s^T tiles (rows = keys, cols = queries) from LDS
    const bf16* kp = kt + lr * (K_ROWB / 2) + lh * 16;
    v8f st0 = {}, st1 = {};
    st0 = wmma_bf(afrag(kp), qf0, st0);
    st0 = wmma_bf(afrag(kp + 32), qf1, st0);
    st1 = wmma_bf(afrag(kp + 16 * (K_ROWB / 2)), qf0, st1);
    st1 = wmma_bf(afrag(kp + 16 * (K_ROWB / 2) + 32), qf1, st1);

    // online softmax in exp2 domain
    float p0[8], p1[8];
    float bm = -1e30f;
#pragma unroll
    for (int i = 0; i < 8; ++i) {
      p0[i] = st0[i] * sc;
      p1[i] = st1[i] * sc;
      bm = fmaxf(bm, fmaxf(p0[i], p1[i]));
    }
    bm = fmaxf(bm, __shfl_xor(bm, 16, 32));
    const float mnew = fmaxf(mrun, bm);
    const float f = __builtin_amdgcn_exp2f(mrun - mnew);
    mrun = mnew;
    lrun *= f;
    o0 *= f; o1 *= f; o2 *= f; o3 *= f;
    float ls = 0.f;
#pragma unroll
    for (int i = 0; i < 8; ++i) {
      p0[i] = __builtin_amdgcn_exp2f(p0[i] - mnew);
      p1[i] = __builtin_amdgcn_exp2f(p1[i] - mnew);
      ls += p0[i] + p1[i];
    }
    lrun += ls;

    // p^T B-fragment: lane-half swap via shfl_xor(16)
    v16bf pf;
#pragma unroll
    for (int i = 0; i < 8; ++i) {
      const float tlo = __shfl_xor(p0[i], 16, 32);
      const float thi = __shfl_xor(p1[i], 16, 32);
      pf[i]     = (bf16)(lh ? thi : p0[i]);
      pf[8 + i] = (bf16)(lh ? p1[i] : tlo);
    }

    // o^T += V^T(d x j) @ p^T(j x m) from LDS
    const bf16* vp = vt + lr * (V_ROWB / 2) + lh * 16;
    o0 = wmma_bf(afrag(vp), pf, o0);
    o1 = wmma_bf(afrag(vp + 16 * (V_ROWB / 2)), pf, o1);
    o2 = wmma_bf(afrag(vp + 32 * (V_ROWB / 2)), pf, o2);
    o3 = wmma_bf(afrag(vp + 48 * (V_ROWB / 2)), pf, o3);

    wait_async0();   // our prefetch into buf^1 landed
    wait_ds0();      // our reads of buf finished
    __syncthreads(); // everyone done -> safe to overwrite buf next iter
    cur ^= 1;
  }

  const float lt = lrun + __shfl_xor(lrun, 16, 32);
  const float inv = 1.0f / lt;
  o0 *= inv; o1 *= inv; o2 *= inv; o3 *= inv;

  // Transpose o^T(d x m) -> [token][d] via wave-private LDS, then coalesced
  // 128B-per-token bf16 stores into the [b,s,D] activation buffer.
  bf16* lb = (bf16*)(smem + 2 * BUF_B) + wid * (16 * HD);
  const v8f oo[4] = {o0, o1, o2, o3};
#pragma unroll
  for (int t = 0; t < 4; ++t)
#pragma unroll
    for (int i = 0; i < 8; ++i)
      lb[lr * HD + t * 16 + i + lh * 8] = (bf16)oo[t][i];

  bf16* gout = O + (b * SEQ + m0) * DIM + h * HD;
#pragma unroll
  for (int pass = 0; pass < 4; ++pass) {
    const int e = pass * 256 + lane * 8;
    const int t = e >> 6, off = e & 63;
    *(v8bf*)(gout + t * DIM + off) = *(const v8bf*)(lb + e);
  }
}

// ---------------------------------------------------------------- launcher
extern "C" void kernel_launch(void* const* d_in, const int* in_sizes, int n_in,
                              void* d_out, int out_size, void* d_ws, size_t ws_size,
                              hipStream_t stream) {
  (void)in_sizes; (void)n_in; (void)out_size; (void)ws_size;
  const float* x  = (const float*)d_in[0];
  const float* Wq = (const float*)d_in[1];
  const float* bq = (const float*)d_in[2];
  const float* Wk = (const float*)d_in[3];
  const float* bk = (const float*)d_in[4];
  const float* Wv = (const float*)d_in[5];
  const float* bv = (const float*)d_in[6];
  const float* Wo = (const float*)d_in[7];
  const float* bo = (const float*)d_in[8];
  float* out = (float*)d_out;

  char* ws = (char*)d_ws;                       // 48 MiB used
  bf16* xb  = (bf16*)(ws);
  bf16* wqb = (bf16*)(ws + (8ull  << 20));
  bf16* wkb = (bf16*)(ws + (10ull << 20));
  bf16* wvb = (bf16*)(ws + (12ull << 20));
  bf16* wob = (bf16*)(ws + (14ull << 20));
  bf16* qb  = (bf16*)(ws + (16ull << 20));      // [b,h,s,d]
  bf16* kb  = (bf16*)(ws + (24ull << 20));      // [b,h,s,d]
  bf16* vtb = (bf16*)(ws + (32ull << 20));      // [b,h,d,s]
  bf16* ab  = (bf16*)(ws + (40ull << 20));      // attn out bf16 [b,s,D]

  cvt_bf16<<<NTOK * DIM / 1024, 256, 0, stream>>>(x, xb, NTOK * DIM);
  cvt_bf16<<<DIM * DIM / 1024, 256, 0, stream>>>(Wq, wqb, DIM * DIM);
  cvt_bf16<<<DIM * DIM / 1024, 256, 0, stream>>>(Wk, wkb, DIM * DIM);
  cvt_bf16<<<DIM * DIM / 1024, 256, 0, stream>>>(Wv, wvb, DIM * DIM);
  cvt_bf16<<<DIM * DIM / 1024, 256, 0, stream>>>(Wo, wob, DIM * DIM);

  const dim3 gg(NTOK / 64, DIM / 256);
  gemm_proj<<<gg, 256, 0, stream>>>(xb, wqb, bq, qb, 0);
  gemm_proj<<<gg, 256, 0, stream>>>(xb, wkb, bk, kb, 0);
  gemm_proj<<<gg, 256, 0, stream>>>(xb, wvb, bv, vtb, 1);

  attn_fwd<<<dim3(SEQ / 128, NH, 2), 256, 0, stream>>>(qb, kb, vtb, ab);

  gemm_out<<<gg, 256, 0, stream>>>(ab, wob, bo, out);
}
